// LSTM_79285096284795
// MI455X (gfx1250) — matrix operand order repaired and checked by reading, not compiled, not run on hardware
//
#include <hip/hip_runtime.h>
#include <hip/hip_bf16.h>

// Persistent 2-layer LSTM (H=64) for MI455X / gfx1250 (wave32, WMMA).
// Grid: 64 WGs x 256 threads; each WG owns a 16-row batch tile for all 512 steps.
// Weights held in per-lane WMMA B-fragments (registers); h/c states in LDS.
// v2.1: exp2-based activations (v_exp_f32 + v_rcp_f32, no IEEE-div expansion),
//       native v_tanh_f32 when available; h2-dependent WMMAs hoisted ahead of
//       the first barrier to shorten the serial recurrence path.

#define B_TOT 1024
#define L_SEQ 512
#define HID   64
#define BT    16          // batch rows per workgroup (WMMA M)
#define HSTR  72          // padded _Float16 stride for h buffers (144B = 36 banks, conflict-free)
#define GSTR  260         // padded float stride for the 16x256 gate buffer

typedef __attribute__((ext_vector_type(16))) _Float16 v16h;
typedef __attribute__((ext_vector_type(8)))  float    v8f;

#define WMMA_F16(A, Bf, C) \
  __builtin_amdgcn_wmma_f32_16x16x32_f16(false, (A), false, (Bf), (short)0, (C), false, false)

#define LOG2E  1.4426950408889634f

// sigmoid via v_exp_f32 (base-2) + v_rcp_f32: no IEEE-div expansion
__device__ __forceinline__ float sigm_f(float x) {
  return __builtin_amdgcn_rcpf(1.0f + __builtin_amdgcn_exp2f(-x * LOG2E));
}

__device__ __forceinline__ float tanh_f(float x) {
#if __has_builtin(__builtin_amdgcn_tanhf)
  return __builtin_amdgcn_tanhf(x);           // native V_TANH_F32 transcendental
#else
  // tanh(x) = 1 - 2 / (e^{2x} + 1), exp2-based, rcp instead of div
  float e = __builtin_amdgcn_exp2f(x * (2.0f * LOG2E));
  return 1.0f - 2.0f * __builtin_amdgcn_rcpf(e + 1.0f);
#endif
}

__global__ __launch_bounds__(256, 1)
void lstm2_persistent(const float* __restrict__ y,
                      const float* __restrict__ W_ih1,
                      const float* __restrict__ W_hh1,
                      const float* __restrict__ b_ih1,
                      const float* __restrict__ b_hh1,
                      const float* __restrict__ W_ih2,
                      const float* __restrict__ W_hh2,
                      const float* __restrict__ b_ih2,
                      const float* __restrict__ b_hh2,
                      const float* __restrict__ W_lin,
                      const float* __restrict__ b_lin,
                      float* __restrict__ out)
{
  __shared__ __align__(16) float    ybuf[BT * L_SEQ];   // staged inputs, 32 KB
  __shared__ __align__(16) float    gbuf[BT * GSTR];    // pre-activation gates, 16.6 KB
  __shared__ __align__(16) _Float16 h1buf[BT * HSTR];   // h1 state (f16, A-operand layout)
  __shared__ __align__(16) _Float16 h2buf[BT * HSTR];   // h2 state
  __shared__ __align__(16) float    c1buf[BT * HID];    // c1 state (f32)
  __shared__ __align__(16) float    c2buf[BT * HID];    // c2 state
  __shared__ __align__(16) float    wlin_s[HID];
  __shared__ __align__(16) float    ps[BT * 16];        // projection partials

  const int tid  = threadIdx.x;
  const int lane = tid & 31;
  const int bb   = blockIdx.x * BT;

  // ---- prologue: stage y tile (coalesced), zero states ----
  for (int idx = tid; idx < BT * L_SEQ; idx += 256) {
    int r = idx >> 9, c = idx & (L_SEQ - 1);
    ybuf[idx] = y[(bb + r) * L_SEQ + c];
  }
  for (int idx = tid; idx < BT * HID;  idx += 256) { c1buf[idx] = 0.f; c2buf[idx] = 0.f; }
  for (int idx = tid; idx < BT * HSTR; idx += 256) {
    h1buf[idx] = (_Float16)0.f; h2buf[idx] = (_Float16)0.f;
  }
  if (tid < HID) wlin_s[tid] = W_lin[tid];

  // ---- per-lane WMMA fragment coordinates (ISA 16x16x32 f16 layouts) ----
  const int lo = lane & 15;      // A: row M / B: col N / C: col N
  const int hi = lane >> 4;      // selects K-half / C row group
  const int wv = tid >> 5;       // wave 0..7 -> owns N-tiles 2*wv, 2*wv+1
  const int n0 = wv * 32 + lo;   // gate column, tile 0
  const int n1 = n0 + 16;        // gate column, tile 1

  // B-fragment loader: W is [256][64] row-major; B[k][n] = W[n][k].
  // lane element t  ->  K = kc*32 + hi*16 + t, N = lo (tile column n)
  auto ldW = [&](const float* W, int n, int kc) {
    v16h v;
    const float* p = W + n * HID + kc * 32 + hi * 16;
#pragma unroll
    for (int t = 0; t < 16; ++t) v[t] = (_Float16)p[t];
    return v;
  };

  v16h Bw1[2][2], Bw2i[2][2], Bw2h[2][2];   // 12 x v16h = 96 VGPRs, persistent
#pragma unroll
  for (int kc = 0; kc < 2; ++kc) {
    Bw1[0][kc]  = ldW(W_hh1, n0, kc);  Bw1[1][kc]  = ldW(W_hh1, n1, kc);
    Bw2i[0][kc] = ldW(W_ih2, n0, kc);  Bw2i[1][kc] = ldW(W_ih2, n1, kc);
    Bw2h[0][kc] = ldW(W_hh2, n0, kc);  Bw2h[1][kc] = ldW(W_hh2, n1, kc);
  }
  const float wih1_0 = W_ih1[n0],             wih1_1 = W_ih1[n1];
  const float bs1_0  = b_ih1[n0] + b_hh1[n0], bs1_1  = b_ih1[n1] + b_hh1[n1];
  const float bs2_0  = b_ih2[n0] + b_hh2[n0], bs2_1  = b_ih2[n1] + b_hh2[n1];
  const float blin   = b_lin[0];

  // A-fragment loader from LDS h buffer (two aligned 16B chunks -> ds_load_b128 x2)
  auto ldA = [&](const _Float16* hb, int kc) {
    const _Float16* p = hb + lo * HSTR + kc * 32 + hi * 16;
    v16h a;
#pragma unroll
    for (int t = 0; t < 8; ++t) { a[t] = p[t]; a[t + 8] = p[t + 8]; }
    return a;
  };

  __syncthreads();

  for (int t = 0; t < L_SEQ; ++t) {
    // ===== phase A: layer-1 gates  AND  the h2-dependent half of layer-2 =====
    // (h2buf holds h2_{t-1} and is not rewritten until after barrier 3, so the
    //  h2 @ W_hh2^T WMMAs can run here, overlapping the elementwise-1 barrier)
    v16h a0 = ldA(h1buf, 0);
    v16h a1 = ldA(h1buf, 1);
    v16h q0 = ldA(h2buf, 0);
    v16h q1 = ldA(h2buf, 1);
    float xv[8];
#pragma unroll
    for (int r = 0; r < 8; ++r) xv[r] = ybuf[(r + 8 * hi) * L_SEQ + t];

    v8f acc0 = {};
    acc0 = WMMA_F16(a0, Bw1[0][0], acc0);
    acc0 = WMMA_F16(a1, Bw1[0][1], acc0);
    v8f acc1 = {};
    acc1 = WMMA_F16(a0, Bw1[1][0], acc1);
    acc1 = WMMA_F16(a1, Bw1[1][1], acc1);

    v8f d0 = {};
    d0 = WMMA_F16(q0, Bw2h[0][0], d0);
    d0 = WMMA_F16(q1, Bw2h[0][1], d0);
    v8f d1 = {};
    d1 = WMMA_F16(q0, Bw2h[1][0], d1);
    d1 = WMMA_F16(q1, Bw2h[1][1], d1);

#pragma unroll
    for (int r = 0; r < 8; ++r) {
      int row = r + 8 * hi;
      gbuf[row * GSTR + n0] = acc0[r] + xv[r] * wih1_0 + bs1_0;
      gbuf[row * GSTR + n1] = acc1[r] + xv[r] * wih1_1 + bs1_1;
    }
    __syncthreads();

    // ---- elementwise layer 1: c1,h1 update ----
    {
      int b  = tid >> 4;
      int j0 = (tid & 15) * 4;
#pragma unroll
      for (int u = 0; u < 4; ++u) {
        int j = j0 + u;
        float gi = gbuf[b * GSTR + j];
        float gf = gbuf[b * GSTR + j + 64];
        float gg = gbuf[b * GSTR + j + 128];
        float go = gbuf[b * GSTR + j + 192];
        float c  = c1buf[b * HID + j];
        c = sigm_f(gf) * c + sigm_f(gi) * tanh_f(gg);
        float h = sigm_f(go) * tanh_f(c);
        c1buf[b * HID + j]  = c;
        h1buf[b * HSTR + j] = (_Float16)h;
      }
    }
    __syncthreads();

    // ===== phase B: finish layer-2 gates with the fresh h1 =====
    v16h p0 = ldA(h1buf, 0);
    v16h p1 = ldA(h1buf, 1);
    d0 = WMMA_F16(p0, Bw2i[0][0], d0);
    d0 = WMMA_F16(p1, Bw2i[0][1], d0);
    d1 = WMMA_F16(p0, Bw2i[1][0], d1);
    d1 = WMMA_F16(p1, Bw2i[1][1], d1);

#pragma unroll
    for (int r = 0; r < 8; ++r) {
      int row = r + 8 * hi;
      gbuf[row * GSTR + n0] = d0[r] + bs2_0;
      gbuf[row * GSTR + n1] = d1[r] + bs2_1;
    }
    __syncthreads();

    // ---- elementwise layer 2: c2,h2 update + projection partials ----
    {
      int b  = tid >> 4;
      int jq = tid & 15;
      int j0 = jq * 4;
      float part = 0.f;
#pragma unroll
      for (int u = 0; u < 4; ++u) {
        int j = j0 + u;
        float gi = gbuf[b * GSTR + j];
        float gf = gbuf[b * GSTR + j + 64];
        float gg = gbuf[b * GSTR + j + 128];
        float go = gbuf[b * GSTR + j + 192];
        float c  = c2buf[b * HID + j];
        c = sigm_f(gf) * c + sigm_f(gi) * tanh_f(gg);
        float h = sigm_f(go) * tanh_f(c);
        c2buf[b * HID + j]  = c;
        h2buf[b * HSTR + j] = (_Float16)h;
        part += h * wlin_s[j];
      }
      ps[b * 16 + jq] = part;
    }
    __syncthreads();

    // ---- out[b, t] = h2 . w_lin + b_lin ----
    if (tid < BT) {
      float s = blin;
#pragma unroll
      for (int q = 0; q < 16; ++q) s += ps[tid * 16 + q];
      out[(bb + tid) * L_SEQ + t] = s;
    }
  }
}

extern "C" void kernel_launch(void* const* d_in, const int* in_sizes, int n_in,
                              void* d_out, int out_size, void* d_ws, size_t ws_size,
                              hipStream_t stream) {
  (void)in_sizes; (void)n_in; (void)out_size; (void)d_ws; (void)ws_size;
  const float* y     = (const float*)d_in[0];
  const float* W_ih1 = (const float*)d_in[1];
  const float* W_hh1 = (const float*)d_in[2];
  const float* b_ih1 = (const float*)d_in[3];
  const float* b_hh1 = (const float*)d_in[4];
  const float* W_ih2 = (const float*)d_in[5];
  const float* W_hh2 = (const float*)d_in[6];
  const float* b_ih2 = (const float*)d_in[7];
  const float* b_hh2 = (const float*)d_in[8];
  const float* W_lin = (const float*)d_in[9];
  const float* b_lin = (const float*)d_in[10];
  float* out = (float*)d_out;

  dim3 grid(B_TOT / BT), block(256);
  lstm2_persistent<<<grid, block, 0, stream>>>(y, W_ih1, W_hh1, b_ih1, b_hh1,
                                               W_ih2, W_hh2, b_ih2, b_hh2,
                                               W_lin, b_lin, out);
}
